// EmformerBlock_20796231647928
// MI455X (gfx1250) — matrix operand order repaired and checked by reading, not compiled, not run on hardware
//
#include <hip/hip_runtime.h>
#include <hip/hip_bf16.h>

typedef __bf16 bf16;
typedef __attribute__((ext_vector_type(16))) __bf16 v16bf;
typedef __attribute__((ext_vector_type(8)))  float  v8f;
typedef unsigned int v4u  __attribute__((ext_vector_type(4)));
typedef int          v8ii __attribute__((ext_vector_type(8)));
typedef int          v4ii __attribute__((ext_vector_type(4)));

#define B_   2
#define T_   2048
#define D_   1024
#define H_   16
#define DH_  64
#define FFN_ 4096
#define ROWS (B_ * T_)   // 4096
#define MPW  (T_ / 32)   // packed mask dwords per row = 64

#if defined(__has_builtin)
#if __has_builtin(__builtin_amdgcn_tensor_load_to_lds) && \
    __has_builtin(__builtin_amdgcn_s_wait_tensorcnt)
#define HAVE_TDM 1
#endif
#endif

__device__ __forceinline__ v8f wmma_bf16(const v16bf& a, const v16bf& b, v8f c) {
    return __builtin_amdgcn_wmma_f32_16x16x32_bf16(
        false, a, false, b, (short)0, c, false, false);
}

#ifdef HAVE_TDM
// 2D TDM load: tile_k contiguous bf16 per row, tile_rows rows, row stride in elems.
// Pads LDS rows: pad_interval_code (2^(c+1) dwords), pad_amount_code (c+1 dwords).
__device__ __forceinline__ void tdm_load_2d(unsigned lds_addr, const void* gptr,
                                            unsigned tile_k, unsigned tile_rows,
                                            unsigned stride_elems,
                                            unsigned pad_int_c, unsigned pad_amt_c) {
    unsigned long long ga = (unsigned long long)gptr;
    v4u g0;
    g0[0] = 1u;                                                  // count=1 (valid D#)
    g0[1] = lds_addr;                                            // lds_addr [63:32]
    g0[2] = (unsigned)ga;                                        // global_addr lo
    g0[3] = (unsigned)((ga >> 32) & 0x01FFFFFFull) | (2u << 30); // addr[56:32] | type=2
    v8ii g1;
    g1[0] = (int)((1u << 16) | (1u << 20) |                      // data_size=2B, pad_en
                  (pad_int_c << 22) | (pad_amt_c << 25));
    g1[1] = (int)((stride_elems & 0xFFFFu) << 16);               // tensor_dim0 lo16
    g1[2] = (int)((stride_elems >> 16) | (0xFFFFu << 16));       // dim0 hi | dim1 lo
    g1[3] = (int)(0x000Fu | ((tile_k & 0xFFFFu) << 16));         // dim1 hi | tile_dim0
    g1[4] = (int)(tile_rows & 0xFFFFu);                          // tile_dim1, tile_dim2=0
    g1[5] = (int)stride_elems;                                   // tensor_dim0_stride lo
    g1[6] = 0;                                                   // stride0 hi | stride1 lo
    g1[7] = 0;
    v4ii z4 = {0, 0, 0, 0};
    v8ii z8 = {0, 0, 0, 0, 0, 0, 0, 0};
    __builtin_amdgcn_tensor_load_to_lds(g0, g1, z4, z4, z8, 0);
}
#endif

// Stage a [rows x 64] bf16 tile (row stride ld elems) into padded LDS [rows][72].
template<int ROWS_T>
__device__ __forceinline__ void stage_tile(bf16 (*dst)[72], const bf16* src,
                                           int ld, int tid) {
#ifdef HAVE_TDM
    if (tid < 32)   // one wave issues the DMA (TENSORcnt tracked)
        tdm_load_2d((unsigned)(size_t)(void*)&dst[0][0], (const void*)src,
                    64u, (unsigned)ROWS_T, (unsigned)ld, 4u, 3u); // 32dw rows, 4dw pad
#else
    if (tid < ROWS_T) {
        const uint4* s = (const uint4*)(src + (size_t)tid * ld);
        uint4* d = (uint4*)&dst[tid][0];
        #pragma unroll
        for (int i = 0; i < 8; ++i) d[i] = s[i];
    }
#endif
}

// Wait for staged tiles: 'more'=true allows 2 outstanding (next prefetch pair),
// false drains all. s_wait_tensorcnt requires a literal immediate.
__device__ __forceinline__ void stage_wait(bool more, int tid) {
#ifdef HAVE_TDM
    if (tid < 32) {
        if (more) __builtin_amdgcn_s_wait_tensorcnt(2);
        else      __builtin_amdgcn_s_wait_tensorcnt(0);
    }
#else
    (void)more; (void)tid;
#endif
    __syncthreads();
}

// ---------------------------------------------------------------------------
// Transpose + convert: src fp32 [Kd][Nd] -> dst bf16 [Nd][Kd]  (32x32 LDS tiles)
// ---------------------------------------------------------------------------
__global__ void __launch_bounds__(256)
transpose_cvt(const float* __restrict__ src, bf16* __restrict__ dst, int Kd, int Nd) {
    __shared__ float t[32][33];
    const int kb = blockIdx.y * 32, nb = blockIdx.x * 32;
    const int tx = threadIdx.x & 31, ty = threadIdx.x >> 5;   // 32 x 8
    #pragma unroll
    for (int i = 0; i < 32; i += 8)
        t[ty + i][tx] = src[(size_t)(kb + ty + i) * Nd + nb + tx];
    __syncthreads();
    #pragma unroll
    for (int i = 0; i < 32; i += 8)
        dst[(size_t)(nb + ty + i) * Kd + kb + tx] = (bf16)t[tx][ty + i];
}

__global__ void pack_qkv_b(const float* __restrict__ bq, const float* __restrict__ bk,
                           const float* __restrict__ bv, float* __restrict__ bqkv) {
    int i = blockIdx.x * blockDim.x + threadIdx.x;
    if (i < D_) { bqkv[i] = bq[i]; bqkv[i + D_] = bk[i]; bqkv[i + 2 * D_] = bv[i]; }
}

// ---------------------------------------------------------------------------
// Bit-pack mask [B,1,T,T] int32 -> [B*T, T/32] dwords via wave32 ballot
// ---------------------------------------------------------------------------
__global__ void __launch_bounds__(256)
pack_mask(const int* __restrict__ mask, unsigned* __restrict__ packed) {
    const int row = blockIdx.x;
    const int tid = threadIdx.x;
    const int* mrow = mask + (size_t)row * T_;
    unsigned* prow = packed + (size_t)row * MPW;
    #pragma unroll
    for (int it = 0; it < T_ / 256; ++it) {
        const int key = it * 256 + tid;
        unsigned long long bal = __ballot(mrow[key] != 0);
        if ((tid & 31) == 0) prow[it * 8 + (tid >> 5)] = (unsigned)bal;
    }
}

// ---------------------------------------------------------------------------
// LayerNorm (D = 1024, one block per row, 256 threads, 4 elems/thread)
// ---------------------------------------------------------------------------
template<bool ADD, bool WRITE_SUM, bool NORM_BF16>
__global__ void __launch_bounds__(256)
ln_kernel(const float* __restrict__ X, const float* __restrict__ R,
          const float* __restrict__ g, const float* __restrict__ bt,
          float* __restrict__ sumout, bf16* __restrict__ nb, float* __restrict__ nf) {
    const int row  = blockIdx.x;
    const int tid  = threadIdx.x;
    const int wave = tid >> 5, lane = tid & 31;

    float4 v = ((const float4*)(X + (size_t)row * D_))[tid];
    if (ADD) {
        float4 rv = ((const float4*)(R + (size_t)row * D_))[tid];
        v.x += rv.x; v.y += rv.y; v.z += rv.z; v.w += rv.w;
    }
    if (WRITE_SUM) ((float4*)(sumout + (size_t)row * D_))[tid] = v;

    float s1 = v.x + v.y + v.z + v.w;
    float s2 = v.x * v.x + v.y * v.y + v.z * v.z + v.w * v.w;
    #pragma unroll
    for (int m = 1; m < 32; m <<= 1) {
        s1 += __shfl_xor(s1, m, 32);
        s2 += __shfl_xor(s2, m, 32);
    }
    __shared__ float r1[8], r2[8];
    if (lane == 0) { r1[wave] = s1; r2[wave] = s2; }
    __syncthreads();
    s1 = 0.f; s2 = 0.f;
    #pragma unroll
    for (int i = 0; i < 8; ++i) { s1 += r1[i]; s2 += r2[i]; }

    const float mu  = s1 * (1.0f / D_);
    const float var = s2 * (1.0f / D_) - mu * mu;
    const float rs  = rsqrtf(var + 1e-3f);

    float4 gv = ((const float4*)g)[tid];
    float4 bv = ((const float4*)bt)[tid];
    float y0 = (v.x - mu) * rs * gv.x + bv.x;
    float y1 = (v.y - mu) * rs * gv.y + bv.y;
    float y2 = (v.z - mu) * rs * gv.z + bv.z;
    float y3 = (v.w - mu) * rs * gv.w + bv.w;
    const int c = tid * 4;
    if (NORM_BF16) {
        bf16* o = nb + (size_t)row * D_ + c;
        o[0] = (bf16)y0; o[1] = (bf16)y1; o[2] = (bf16)y2; o[3] = (bf16)y3;
    } else {
        ((float4*)(nf + (size_t)row * D_))[tid] = make_float4(y0, y1, y2, y3);
    }
}

// ---------------------------------------------------------------------------
// WMMA bf16 GEMM: C[M,N] = A[M,K] * Bt[N,K]^T + bias (+ReLU) (+resid)
// Block 128 thr (4 waves), tile 128x128, BK=64, TDM double-buffered staging.
// ---------------------------------------------------------------------------
template<bool RELU, bool BF16OUT, bool RESID>
__global__ void __launch_bounds__(128)
gemm_kernel(const bf16* __restrict__ A, int lda,
            const bf16* __restrict__ Bt, int ldb,     // Bt = W^T [N][K]
            const float* __restrict__ bias,
            const float* __restrict__ resid,
            bf16* __restrict__ Cb, float* __restrict__ Cf, int ldc,
            int M, int N, int K) {
    __shared__ bf16 Ash[2][128][72];   // [buf][m][k], 144B rows (16B aligned)
    __shared__ bf16 Bsh[2][128][72];   // [buf][n][k]

    const int tid  = threadIdx.x;
    const int wave = tid >> 5, lane = tid & 31, hi = lane >> 4, lo = lane & 15;
    const int m0 = blockIdx.y * 128, n0 = blockIdx.x * 128;
    const int wm = (wave & 1) * 64, wn = (wave >> 1) * 64;

    v8f acc[4][4] = {};

    const int nk = K >> 6;   // 64-wide K steps
    stage_tile<128>(Ash[0], A  + (size_t)m0 * lda, lda, tid);
    stage_tile<128>(Bsh[0], Bt + (size_t)n0 * ldb, ldb, tid);

    for (int i = 0; i < nk; ++i) {
        const int cb = i & 1;
        const bool more = (i + 1 < nk);
        if (more) {
            stage_tile<128>(Ash[cb ^ 1], A  + (size_t)m0 * lda + (i + 1) * 64, lda, tid);
            stage_tile<128>(Bsh[cb ^ 1], Bt + (size_t)n0 * ldb + (i + 1) * 64, ldb, tid);
        }
        stage_wait(more, tid);   // oldest pair complete (+barrier)

        #pragma unroll
        for (int ks = 0; ks < 2; ++ks) {
            v16bf af[4], bfr[4];
            #pragma unroll
            for (int ii = 0; ii < 4; ++ii) {     // A frags: lane=M, K split by half
                const bf16* ar = &Ash[cb][wm + ii * 16 + lo][ks * 32];
                #pragma unroll
                for (int e = 0; e < 8; ++e) {
                    af[ii][e]     = ar[hi * 16 + e];
                    af[ii][8 + e] = ar[hi * 16 + 8 + e];
                }
            }
            #pragma unroll
            for (int j = 0; j < 4; ++j) {        // B frags: lane=N, 16 contiguous K
                const bf16* br = &Bsh[cb][wn + j * 16 + lo][ks * 32 + hi * 16];
                #pragma unroll
                for (int e = 0; e < 16; ++e) bfr[j][e] = br[e];
            }
            #pragma unroll
            for (int ii = 0; ii < 4; ++ii)
                #pragma unroll
                for (int j = 0; j < 4; ++j)
                    acc[ii][j] = wmma_bf16(af[ii], bfr[j], acc[ii][j]);
        }
        __syncthreads();
    }

    #pragma unroll
    for (int i = 0; i < 4; ++i)
        #pragma unroll
        for (int j = 0; j < 4; ++j) {
            const int col = n0 + wn + j * 16 + lo;
            const float bv = bias[col];
            #pragma unroll
            for (int r = 0; r < 8; ++r) {
                const int row = m0 + wm + i * 16 + r + hi * 8;
                float v = acc[i][j][r] + bv;
                if (RELU)  v = fmaxf(v, 0.f);
                if (RESID) v += resid[(size_t)row * ldc + col];
                if (BF16OUT) Cb[(size_t)row * ldc + col] = (bf16)v;
                else         Cf[(size_t)row * ldc + col] = v;
            }
        }
}

// ---------------------------------------------------------------------------
// Flash attention, transposed scores (S^T = K*Q^T) so P frags are in-register.
// K tile staged via TDM; V staged transposed manually; mask from packed bits.
// ---------------------------------------------------------------------------
__global__ void __launch_bounds__(128)
attn_kernel(const bf16* __restrict__ qkv, const unsigned* __restrict__ mpack,
            float* __restrict__ attn) {
    const int qb = blockIdx.x;
    const int b  = blockIdx.y >> 4;     // H_ = 16
    const int h  = blockIdx.y & 15;
    const int tid = threadIdx.x;
    const int wave = tid >> 5, lane = tid & 31, hi = lane >> 4, lo = lane & 15;
    const size_t ldq = 3 * D_;

    __shared__ bf16 Ksh[128][72];         // [key][d]  (TDM-padded rows)
    __shared__ bf16 VshT[DH_][128 + 8];   // [d][key]  transposed
    __shared__ uint4 Msh[128];            // packed mask bits per qrow

    v16bf qf[2][2];   // B-side of S^T: lane -> col=qrow, 16 contiguous d
    {
        const bf16* qbase = qkv + (size_t)(b * T_ + qb * 128 + wave * 32) * ldq + h * DH_;
        #pragma unroll
        for (int mt = 0; mt < 2; ++mt)
            #pragma unroll
            for (int ks = 0; ks < 2; ++ks) {
                const bf16* qr = qbase + (size_t)(mt * 16 + lo) * ldq + ks * 32 + hi * 16;
                #pragma unroll
                for (int e = 0; e < 16; ++e) qf[mt][ks][e] = qr[e];
            }
    }

    float rowm[2], rowl[2];
    rowm[0] = rowm[1] = -3.0e38f;
    rowl[0] = rowl[1] = 0.f;
    v8f oacc[2][4] = {};

    for (int kb = 0; kb < T_ / 128; ++kb) {
        // K tile via TDM (or fallback), V transposed + mask bits by all threads
        stage_tile<128>(Ksh, qkv + (size_t)(b * T_ + kb * 128) * ldq + D_ + h * DH_,
                        (int)ldq, tid);
        {
            const int p2 = (tid & 63) * 2;
            const int hd = (tid >> 6) * 32;
            const bf16* v0 = qkv + (size_t)(b * T_ + kb * 128 + p2) * ldq + 2 * D_ + h * DH_ + hd;
            const bf16* v1 = v0 + ldq;
            union U { uint4 q[4]; unsigned short s[32]; } a0, a1;
            #pragma unroll
            for (int i = 0; i < 4; ++i) { a0.q[i] = ((const uint4*)v0)[i];
                                          a1.q[i] = ((const uint4*)v1)[i]; }
            #pragma unroll
            for (int d = 0; d < 32; ++d)
                *(unsigned*)&VshT[hd + d][p2] =
                    (unsigned)a0.s[d] | ((unsigned)a1.s[d] << 16);

            Msh[tid] = ((const uint4*)(mpack + (size_t)(b * T_ + qb * 128 + tid) * MPW))[kb];
        }
        stage_wait(false, tid);

        // S^T = K * Q^T
        v8f sacc[8][2] = {};
        #pragma unroll
        for (int ks = 0; ks < 2; ++ks) {
            v16bf kf[8];
            #pragma unroll
            for (int nt = 0; nt < 8; ++nt) {
                const bf16* kr = &Ksh[nt * 16 + lo][ks * 32];
                #pragma unroll
                for (int e = 0; e < 8; ++e) {
                    kf[nt][e]     = kr[hi * 16 + e];
                    kf[nt][8 + e] = kr[hi * 16 + 8 + e];
                }
            }
            #pragma unroll
            for (int nt = 0; nt < 8; ++nt)
                #pragma unroll
                for (int mt = 0; mt < 2; ++mt)
                    sacc[nt][mt] = wmma_bf16(kf[nt], qf[mt][ks], sacc[nt][mt]);
        }

        // scale + packed mask + online softmax
        #pragma unroll
        for (int mt = 0; mt < 2; ++mt) {
            const uint4 mq = Msh[wave * 32 + mt * 16 + lo];
            const unsigned mwa[4] = {mq.x, mq.y, mq.z, mq.w};
            float bmax = -3.0e38f;
            #pragma unroll
            for (int nt = 0; nt < 8; ++nt)
                #pragma unroll
                for (int r = 0; r < 8; ++r) {
                    const unsigned bit = nt * 16 + r + 8 * hi;
                    const unsigned mv = (mwa[bit >> 5] >> (bit & 31)) & 1u;
                    float s = sacc[nt][mt][r] * 0.125f + (mv ? 0.f : -1.0e9f);
                    sacc[nt][mt][r] = s;
                    bmax = fmaxf(bmax, s);
                }
            bmax = fmaxf(bmax, __shfl_xor(bmax, 16, 32));
            const float mnew  = fmaxf(rowm[mt], bmax);
            const float alpha = __expf(rowm[mt] - mnew);
            rowm[mt] = mnew;
            float lsum = 0.f;
            #pragma unroll
            for (int nt = 0; nt < 8; ++nt)
                #pragma unroll
                for (int r = 0; r < 8; ++r) {
                    const float s = sacc[nt][mt][r];
                    const float pden = __expf(s - mnew);
                    lsum += pden;
                    sacc[nt][mt][r] = (s > -1.0e8f) ? pden : 0.f;
                }
            lsum += __shfl_xor(lsum, 16, 32);
            rowl[mt] = rowl[mt] * alpha + lsum;
            float arr[8];
            #pragma unroll
            for (int r = 0; r < 8; ++r) arr[r] = __shfl(alpha, r + 8 * hi, 32);
            #pragma unroll
            for (int dt = 0; dt < 4; ++dt)
                #pragma unroll
                for (int r = 0; r < 8; ++r) oacc[mt][dt][r] *= arr[r];
        }

        // O += P * V : P frags straight from registers
        #pragma unroll
        for (int kp = 0; kp < 4; ++kp) {
            v16bf pf[2];
            #pragma unroll
            for (int mt = 0; mt < 2; ++mt)
                #pragma unroll
                for (int e = 0; e < 8; ++e) {
                    pf[mt][e]     = (bf16)sacc[2 * kp][mt][e];
                    pf[mt][8 + e] = (bf16)sacc[2 * kp + 1][mt][e];
                }
            #pragma unroll
            for (int dt = 0; dt < 4; ++dt) {
                v16bf vf;
                const bf16* vr = &VshT[dt * 16 + lo][kp * 32 + hi * 8];
                #pragma unroll
                for (int e = 0; e < 16; ++e) vf[e] = vr[(e & 8) * 2 + (e & 7)];
                #pragma unroll
                for (int mt = 0; mt < 2; ++mt)
                    oacc[mt][dt] = wmma_bf16(pf[mt], vf, oacc[mt][dt]);
            }
        }
        __syncthreads();
    }

    #pragma unroll
    for (int mt = 0; mt < 2; ++mt) {
        float rlv[8];
        #pragma unroll
        for (int r = 0; r < 8; ++r) rlv[r] = __shfl(rowl[mt], r + 8 * hi, 32);
        #pragma unroll
        for (int dt = 0; dt < 4; ++dt)
            #pragma unroll
            for (int r = 0; r < 8; ++r) {
                const int qrow = qb * 128 + wave * 32 + mt * 16 + r + 8 * hi;
                attn[(size_t)(b * T_ + qrow) * D_ + h * DH_ + dt * 16 + lo] =
                    oacc[mt][dt][r] / rlv[r];
            }
    }
}

// ---------------------------------------------------------------------------
// Host launcher
// ---------------------------------------------------------------------------
extern "C" void kernel_launch(void* const* d_in, const int* in_sizes, int n_in,
                              void* d_out, int out_size, void* d_ws, size_t ws_size,
                              hipStream_t stream) {
    (void)in_sizes; (void)n_in; (void)out_size; (void)ws_size;
    const float* x    = (const float*)d_in[0];
    const int*   mask = (const int*)  d_in[1];
    const float* Wq   = (const float*)d_in[2];
    const float* bq   = (const float*)d_in[3];
    const float* Wk   = (const float*)d_in[4];
    const float* bk   = (const float*)d_in[5];
    const float* Wv   = (const float*)d_in[6];
    const float* bv   = (const float*)d_in[7];
    const float* g_in = (const float*)d_in[8];
    const float* b_in = (const float*)d_in[9];
    const float* g1   = (const float*)d_in[10];
    const float* b1   = (const float*)d_in[11];
    const float* Wff1 = (const float*)d_in[12];
    const float* bff1 = (const float*)d_in[13];
    const float* Wff2 = (const float*)d_in[14];
    const float* bff2 = (const float*)d_in[15];
    const float* g2   = (const float*)d_in[16];
    const float* b2   = (const float*)d_in[17];
    float* out = (float*)d_out;

    char* p = (char*)d_ws;
    auto alloc = [&](size_t bytes) -> void* {
        void* r = (void*)p;
        p += (bytes + 255) & ~(size_t)255;
        return r;
    };
    bf16*     WqkvT = (bf16*)    alloc((size_t)3 * D_ * D_ * 2);   // [3072][1024]
    float*    bqkv  = (float*)   alloc((size_t)3 * D_ * 4);
    bf16*     Wf1T  = (bf16*)    alloc((size_t)FFN_ * D_ * 2);     // [4096][1024]
    bf16*     Wf2T  = (bf16*)    alloc((size_t)D_ * FFN_ * 2);     // [1024][4096]
    bf16*     xn    = (bf16*)    alloc((size_t)ROWS * D_ * 2);
    bf16*     qkv   = (bf16*)    alloc((size_t)ROWS * 3 * D_ * 2);
    float*    attn  = (float*)   alloc((size_t)ROWS * D_ * 4);
    float*    attno = (float*)   alloc((size_t)ROWS * D_ * 4);
    bf16*     h1n   = (bf16*)    alloc((size_t)ROWS * D_ * 2);
    bf16*     h2    = (bf16*)    alloc((size_t)ROWS * FFN_ * 2);
    float*    hpre  = (float*)   alloc((size_t)ROWS * D_ * 4);
    unsigned* mpk   = (unsigned*)alloc((size_t)ROWS * MPW * 4);

    // one-time preprocessing: transpose+convert weights, pack bias + mask bits
    transpose_cvt<<<dim3(D_ / 32, D_ / 32), 256, 0, stream>>>(Wq, WqkvT, D_, D_);
    transpose_cvt<<<dim3(D_ / 32, D_ / 32), 256, 0, stream>>>(Wk, WqkvT + (size_t)D_ * D_, D_, D_);
    transpose_cvt<<<dim3(D_ / 32, D_ / 32), 256, 0, stream>>>(Wv, WqkvT + (size_t)2 * D_ * D_, D_, D_);
    transpose_cvt<<<dim3(FFN_ / 32, D_ / 32), 256, 0, stream>>>(Wff1, Wf1T, D_, FFN_);
    transpose_cvt<<<dim3(D_ / 32, FFN_ / 32), 256, 0, stream>>>(Wff2, Wf2T, FFN_, D_);
    pack_qkv_b<<<(3 * D_ + 255) / 256, 256, 0, stream>>>(bq, bk, bv, bqkv);
    pack_mask<<<ROWS, 256, 0, stream>>>(mask, mpk);

    // pre-LN -> bf16 xn
    ln_kernel<false, false, true><<<ROWS, 256, 0, stream>>>(x, nullptr, g_in, b_in,
                                                            nullptr, xn, nullptr);
    // fused QKV projection: [4096,1024] x [1024,3072] -> bf16 qkv
    gemm_kernel<false, true, false><<<dim3(24, 32), 128, 0, stream>>>(
        xn, D_, WqkvT, D_, bqkv, nullptr, qkv, nullptr, 3 * D_, ROWS, 3 * D_, D_);
    // flash attention -> fp32 attn
    attn_kernel<<<dim3(T_ / 128, B_ * H_), 128, 0, stream>>>(qkv, mpk, attn);
    // attn_out = attn + x (fp32), h1n = LN(attn_out) (bf16)
    ln_kernel<true, true, true><<<ROWS, 256, 0, stream>>>(attn, x, g1, b1,
                                                          attno, h1n, nullptr);
    // FFN1 + ReLU
    gemm_kernel<true, true, false><<<dim3(32, 32), 128, 0, stream>>>(
        h1n, D_, Wf1T, D_, bff1, nullptr, h2, nullptr, FFN_, ROWS, FFN_, D_);
    // FFN2 + bias + residual(attn_out)
    gemm_kernel<false, false, true><<<dim3(8, 32), 128, 0, stream>>>(
        h2, FFN_, Wf2T, FFN_, bff2, attno, nullptr, hpre, D_, ROWS, D_, FFN_);
    // final LN -> fp32 out
    ln_kernel<false, false, false><<<ROWS, 256, 0, stream>>>(hpre, nullptr, g2, b2,
                                                             nullptr, nullptr, out);
}
